// simOTA_23880018165943
// MI455X (gfx1250) — compile-verified, matrix operand order checked by reading, not run on hardware
//
#include <hip/hip_runtime.h>
#include <stdint.h>

#define HW    8400
#define NCLS  80
#define BSZ   16
#define NG    32
#define TOPK  10
#define EPSF  1e-7f
#define IEPS  1e-9f

typedef __attribute__((ext_vector_type(16))) _Float16 v16h;
typedef __attribute__((ext_vector_type(8)))  float    v8f;

// Raw v_log_f32 (log2) + ln2 scale. All call sites have arguments in
// [1e-7, 2], far from denormal/inf, so the precise-log range reduction
// that libm's logf emits (ldexp prescale + compensated fma) is unnecessary.
__device__ __forceinline__ float fast_log(float x) {
  return __builtin_amdgcn_logf(x) * 0.693147180559945f;
}

// ---------------------------------------------------------------------------
// K1: per-anchor class sums S = sum_c -max(log(1-p),-100), Se = sum_c -log(1-p+eps)
// One wave per 16-anchor tile. pred_scores tile (16x80 f32, contiguous 5120B)
// is staged to LDS with async b128 loads (padded rows: 84 dwords -> conflict-free
// per-row lane reads), then reduced over the class dim with v_wmma_f32_16x16x32_f16
// against an all-ones B (f32 accumulation; K-layout irrelevant for ones-B).
// ---------------------------------------------------------------------------
__global__ void __launch_bounds__(256) k1_rowsums(const float* __restrict__ ps,
                                                  float* __restrict__ S,
                                                  float* __restrict__ Se) {
  __shared__ float sh[8 * 16 * 84];            // 8 waves x 16 rows x 84 (pad) = 43008 B
  const int lane = threadIdx.x & 31;
  const int wave = threadIdx.x >> 5;
  const int tile = blockIdx.x * 8 + wave;      // 16*525 = 8400 tiles exactly
  const int b    = tile / 525;
  const int a0   = (tile % 525) * 16;

  const float* gsrc = ps + ((size_t)b * HW + a0) * NCLS;   // 1280 contiguous floats
  unsigned ldsBase = (unsigned)(unsigned long long)(uintptr_t)&sh[wave * 16 * 84];
  unsigned long long gbase = (unsigned long long)(uintptr_t)gsrc;

  // 320 x 16B chunks; scatter rows to 336B-padded LDS rows
#pragma unroll
  for (int i = 0; i < 10; ++i) {
    int t = i * 32 + lane;                     // 0..319
    int row = t / 20, seg = t % 20;
    unsigned laddr = ldsBase + (unsigned)(row * 336 + seg * 16);
    unsigned long long gaddr = gbase + (unsigned long long)(t * 16);
    asm volatile("global_load_async_to_lds_b128 %0, %1, off"
                 :: "v"(laddr), "v"(gaddr) : "memory");
  }
  asm volatile("s_wait_asynccnt 0x0" ::: "memory");

  const int r = lane & 15;                     // A-matrix row M = lane % 16
  const int halfsel = lane >> 4;               // K half select
  const float* lrow = &sh[wave * 1344 + r * 84];

  v16h onesB;
#pragma unroll
  for (int j = 0; j < 16; ++j) onesB[j] = (_Float16)1.0f;
  v8f accS = {}, accSe = {};

#pragma unroll
  for (int c = 0; c < 3; ++c) {                // classes 0..95 (80 real + 16 zero pad)
    int colbase = c * 32 + halfsel * 16;
    v16h aS, aSe;
#pragma unroll
    for (int j = 0; j < 16; ++j) {
      int col = colbase + j;
      int cc  = col < NCLS ? col : 0;
      float p  = lrow[cc];
      float om = 1.0f - p;
      float nll1 = -fmaxf(fast_log(om), -100.0f);
      float se   = -fast_log(om + EPSF);
      bool ok = col < NCLS;
      aS[j]  = (_Float16)(ok ? nll1 : 0.0f);
      aSe[j] = (_Float16)(ok ? se   : 0.0f);
    }
    accS  = __builtin_amdgcn_wmma_f32_16x16x32_f16(false, aS,  false, onesB, (short)0, accS,  false, false);
    accSe = __builtin_amdgcn_wmma_f32_16x16x32_f16(false, aSe, false, onesB, (short)0, accSe, false, false);
  }

  // D layout: lanes 0-15 hold rows 0-7 in elems 0-7; lanes 16-31 hold rows 8-15.
#pragma unroll
  for (int j = 0; j < 8; ++j) {
    int row = (lane == j) ? j : ((lane == 16 + j) ? 8 + j : -1);
    if (row >= 0) {
      S [(size_t)b * HW + a0 + row] = accS[j];
      Se[(size_t)b * HW + a0 + row] = accSe[j];
    }
  }
}

// ---------------------------------------------------------------------------
// K2: full (b,g,a) cost + fg-masked IoU
// ---------------------------------------------------------------------------
__global__ void __launch_bounds__(256) k2_cost(
    const float4* __restrict__ pred_boxes, const float4* __restrict__ gt_boxes,
    const unsigned char* __restrict__ mask_gt, const float* __restrict__ ps,
    const int* __restrict__ gt_labels, const float2* __restrict__ anch,
    const float* __restrict__ stridet, const float* __restrict__ S,
    float* __restrict__ cost, float* __restrict__ iouF) {
  const int bg = blockIdx.x;                   // b*32+g
  const int b  = bg >> 5;
  const int a  = blockIdx.y * 256 + threadIdx.x;
  if (a >= HW) return;

  float4 gb = gt_boxes[bg];
  int   lab = gt_labels[bg];
  bool  mg  = mask_gt[bg] != 0;

  float2 ap = anch[a];
  float st  = stridet[a];
  float apx = (ap.x + 0.5f) * st, apy = (ap.y + 0.5f) * st;

  bool in_boxes = (apx > gb.x) && (apy > gb.y) && (gb.z > apx) && (gb.w > apy);
  float cx = (gb.x + gb.z) * 0.5f, cy = (gb.y + gb.w) * 0.5f;
  float dis = 2.5f * st;
  bool in_centers = (apx > cx - dis) && (apy > cy - dis) &&
                    (cx + dis > apx) && (cy + dis > apy);
  bool fg = in_boxes || in_centers;
  bool in_both = in_boxes && in_centers;

  float4 pb = pred_boxes[(size_t)b * HW + a];
  float w1 = pb.z - pb.x, h1 = pb.w - pb.y + IEPS;
  float w2 = gb.z - gb.x, h2 = gb.w - gb.y + IEPS;
  float iw = fmaxf(fminf(pb.z, gb.z) - fmaxf(pb.x, gb.x), 0.0f);
  float ih = fmaxf(fminf(pb.w, gb.w) - fmaxf(pb.y, gb.y), 0.0f);
  float inter = iw * ih;
  float iou = inter / (w1 * h1 + w2 * h2 - inter + IEPS);
  float iou_cost = -fast_log(iou + EPSF);

  float pl = ps[((size_t)b * HW + a) * NCLS + lab];
  float nll0 = -fmaxf(fast_log(pl), -100.0f);
  float nll1 = -fmaxf(fast_log(1.0f - pl), -100.0f);

  float c = S[(size_t)b * HW + a] + (fg ? (nll0 - nll1) : 0.0f)
          + 3.0f * iou_cost + (in_both ? 0.0f : 100000.0f);
  if (!mg) c = 1000000.0f;

  size_t idx = (size_t)bg * HW + a;
  cost[idx] = c;
  iouF[idx] = fg ? iou : 0.0f;
}

// ---------------------------------------------------------------------------
// K3: dynamic-k per (b,g): sum of top-10 fg IoUs -> pt_num in [0,10]
// ---------------------------------------------------------------------------
__global__ void __launch_bounds__(32) k3_ptnum(const float* __restrict__ iouF,
                                               const unsigned char* __restrict__ mask_gt,
                                               int* __restrict__ pt_num) {
  const int bg = blockIdx.x, lane = threadIdx.x;
  const float* row = iouF + (size_t)bg * HW;
  float t[TOPK];
#pragma unroll
  for (int j = 0; j < TOPK; ++j) t[j] = -1.0f;   // t[0] = min of kept
  for (int i = lane; i < HW; i += 32) {
    float v = row[i];
    if (v > t[0]) {
      t[0] = v;
#pragma unroll
      for (int k = 0; k < TOPK - 1; ++k)
        if (t[k] > t[k + 1]) { float tmp = t[k]; t[k] = t[k + 1]; t[k + 1] = tmp; }
    }
  }
  __shared__ float buf[32 * TOPK];
#pragma unroll
  for (int j = 0; j < TOPK; ++j) buf[lane * TOPK + j] = t[j];
  __syncthreads();
  if (lane == 0) {
    float sum = 0.0f;
    for (int k = 0; k < TOPK; ++k) {
      float m = -2.0f; int mp = 0;
      for (int i = 0; i < 32 * TOPK; ++i) if (buf[i] > m) { m = buf[i]; mp = i; }
      buf[mp] = -2.0f;
      sum += fmaxf(m, 0.0f);
    }
    int pn = (int)fmaxf(sum, 1.0f);
    pt_num[bg] = (mask_gt[bg] != 0) ? pn : 0;
  }
}

// ---------------------------------------------------------------------------
// K4: per (b,g): select pt_num smallest costs with stable (value,index) lex
// tie-break (== double-argsort ranks). Mark per-anchor count + selecting g.
// ---------------------------------------------------------------------------
__global__ void __launch_bounds__(32) k4_select(const float* __restrict__ cost,
                                                const int* __restrict__ pt_num,
                                                int* __restrict__ count,
                                                int* __restrict__ g_of) {
  const int bg = blockIdx.x, b = bg >> 5, g = bg & 31, lane = threadIdx.x;
  const int pn = pt_num[bg];
  __shared__ float bv[32 * TOPK];
  __shared__ int   bi[32 * TOPK];
  if (pn <= 0) return;

  const float* row = cost + (size_t)bg * HW;
  float tv[TOPK]; int ti[TOPK];
#pragma unroll
  for (int j = 0; j < TOPK; ++j) { tv[j] = 3.4e38f; ti[j] = 0x7fffffff; }
  for (int i = lane; i < HW; i += 32) {
    float v = row[i];
    if (v < tv[TOPK - 1] || (v == tv[TOPK - 1] && i < ti[TOPK - 1])) {
      tv[TOPK - 1] = v; ti[TOPK - 1] = i;
#pragma unroll
      for (int k = TOPK - 2; k >= 0; --k) {
        bool sw = (tv[k + 1] < tv[k]) || (tv[k + 1] == tv[k] && ti[k + 1] < ti[k]);
        if (sw) {
          float fv = tv[k]; tv[k] = tv[k + 1]; tv[k + 1] = fv;
          int   fi = ti[k]; ti[k] = ti[k + 1]; ti[k + 1] = fi;
        }
      }
    }
  }
#pragma unroll
  for (int j = 0; j < TOPK; ++j) { bv[lane * TOPK + j] = tv[j]; bi[lane * TOPK + j] = ti[j]; }
  __syncthreads();
  if (lane == 0) {
    for (int k = 0; k < pn; ++k) {
      float mv = 3.4e38f; int mi = 0x7fffffff; int mp = -1;
      for (int i = 0; i < 32 * TOPK; ++i)
        if (bv[i] < mv || (bv[i] == mv && bi[i] < mi)) { mv = bv[i]; mi = bi[i]; mp = i; }
      bv[mp] = 3.4e38f; bi[mp] = 0x7fffffff;
      atomicAdd(&count[(size_t)b * HW + mi], 1);
      atomicExch(&g_of[(size_t)b * HW + mi], g);
    }
  }
}

// ---------------------------------------------------------------------------
// K5: per-anchor loss contributions + per-batch partial sums
// ---------------------------------------------------------------------------
__device__ __forceinline__ float ciou_fn(float4 p, float4 g) {
  float w1 = p.z - p.x, h1 = p.w - p.y + IEPS;
  float w2 = g.z - g.x, h2 = g.w - g.y + IEPS;
  float iw = fmaxf(fminf(p.z, g.z) - fmaxf(p.x, g.x), 0.0f);
  float ih = fmaxf(fminf(p.w, g.w) - fmaxf(p.y, g.y), 0.0f);
  float inter = iw * ih;
  float iou = inter / (w1 * h1 + w2 * h2 - inter + IEPS);
  float cw = fmaxf(p.z, g.z) - fminf(p.x, g.x);
  float ch = fmaxf(p.w, g.w) - fminf(p.y, g.y);
  float c2 = cw * cw + ch * ch + IEPS;
  float dx = g.x + g.z - p.x - p.z, dy = g.y + g.w - p.y - p.w;
  float d2 = (dx * dx + dy * dy) * 0.25f;
  float ang = atanf(w2 / h2) - atanf(w1 / h1);
  float v = 0.40528473456935108577f * ang * ang;   // 4/pi^2
  float alpha = v / (v - iou + 1.0f + IEPS);
  return iou - (d2 / c2 + v * alpha);
}

__global__ void __launch_bounds__(256) k5_final(
    const int* __restrict__ count, const int* __restrict__ g_of,
    const float* __restrict__ cost, const float4* __restrict__ pred_boxes,
    const float4* __restrict__ gt_boxes, const int* __restrict__ gt_labels,
    const float* __restrict__ ps, const float* __restrict__ pconf,
    const float* __restrict__ Se, float* __restrict__ sums) {
  const int b = blockIdx.y;
  const int a = blockIdx.x * 256 + threadIdx.x;
  float c0 = 0.0f, c1 = 0.0f, c2 = 0.0f, cn = 0.0f;
  if (a < HW) {
    size_t ba = (size_t)b * HW + a;
    int cnt = count[ba];
    float pc = pconf[ba];
    if (cnt > 0) { c2 = -fast_log(pc + EPSF); cn = 1.0f; }
    else         { c2 = -fast_log(1.0f - pc + EPSF); }
    if (cnt > 0) {
      int gs;
      if (cnt == 1) gs = g_of[ba];
      else {                                        // onehot of argmin over g (first-min)
        float bvv = 3.4e38f; gs = 0;
        for (int g = 0; g < NG; ++g) {
          float v = cost[((size_t)(b * NG + g)) * HW + a];
          if (v < bvv) { bvv = v; gs = g; }
        }
      }
      float4 pb = pred_boxes[ba];
      float4 gbx = gt_boxes[b * NG + gs];
      c0 = 1.0f - ciou_fn(pb, gbx);
      int lab = gt_labels[b * NG + gs];
      float pl = ps[ba * NCLS + lab];
      c1 = Se[ba] - fast_log(pl + EPSF) + fast_log(1.0f - pl + EPSF);
    }
  }
  __shared__ float red[256];
  float vals[4] = {c0, c1, c2, cn};
  for (int k = 0; k < 4; ++k) {
    red[threadIdx.x] = vals[k];
    __syncthreads();
    for (int s = 128; s > 0; s >>= 1) {
      if (threadIdx.x < s) red[threadIdx.x] += red[threadIdx.x + s];
      __syncthreads();
    }
    if (threadIdx.x == 0) atomicAdd(&sums[k * BSZ + b], red[0]);
    __syncthreads();
  }
}

__global__ void k6_out(const float* __restrict__ sums, float* __restrict__ out) {
  if (threadIdx.x == 0) {
    float l0 = 0.0f, l1 = 0.0f, l2 = 0.0f;
    for (int b = 0; b < BSZ; ++b) {
      float np = sums[3 * BSZ + b];
      l0 += sums[b] / np;
      l1 += sums[BSZ + b] / np;
      l2 += sums[2 * BSZ + b] / np;
    }
    out[0] = l0; out[1] = l1; out[2] = l2;
  }
}

// ---------------------------------------------------------------------------
extern "C" void kernel_launch(void* const* d_in, const int* in_sizes, int n_in,
                              void* d_out, int out_size, void* d_ws, size_t ws_size,
                              hipStream_t stream) {
  const float4*        pred_boxes = (const float4*)d_in[0];
  const float4*        gt_boxes   = (const float4*)d_in[1];
  const unsigned char* mask_gt    = (const unsigned char*)d_in[2];
  const float*         ps         = (const float*)d_in[3];
  const float*         pconf      = (const float*)d_in[4];
  const int*           gt_labels  = (const int*)d_in[5];
  const float2*        anch       = (const float2*)d_in[6];
  const float*         stridet    = (const float*)d_in[7];
  float* out = (float*)d_out;

  char* ws = (char*)d_ws;
  float* S      = (float*)(ws);                    // 134400 f
  float* Se     = (float*)(ws + 537600);           // 134400 f
  float* cost   = (float*)(ws + 1075200);          // 4300800 f
  float* iouF   = (float*)(ws + 18278400);         // 4300800 f
  int*   pt_num = (int*)  (ws + 35481600);         // 512 i32
  int*   count  = (int*)  (ws + 35483648);         // 134400 i32
  int*   g_of   = (int*)  (ws + 36021248);         // 134400 i32
  float* sums   = (float*)(ws + 36558848);         // 64 f

  hipMemsetAsync(count, 0, 134400 * sizeof(int), stream);
  hipMemsetAsync(sums, 0, 64 * sizeof(float), stream);

  k1_rowsums<<<1050, 256, 0, stream>>>(ps, S, Se);
  k2_cost<<<dim3(BSZ * NG, (HW + 255) / 256), 256, 0, stream>>>(
      pred_boxes, gt_boxes, mask_gt, ps, gt_labels, anch, stridet, S, cost, iouF);
  k3_ptnum<<<BSZ * NG, 32, 0, stream>>>(iouF, mask_gt, pt_num);
  k4_select<<<BSZ * NG, 32, 0, stream>>>(cost, pt_num, count, g_of);
  k5_final<<<dim3((HW + 255) / 256, BSZ), 256, 0, stream>>>(
      count, g_of, cost, pred_boxes, gt_boxes, gt_labels, ps, pconf, Se, sums);
  k6_out<<<1, 32, 0, stream>>>(sums, out);
}
